// LSTMAttn0_38723425141445
// MI455X (gfx1250) — matrix operand order repaired and checked by reading, not compiled, not run on hardware
//
#include <hip/hip_runtime.h>

typedef __attribute__((ext_vector_type(16))) __bf16 v16bf;
typedef __attribute__((ext_vector_type(8)))  __bf16 v8bf;
typedef __attribute__((ext_vector_type(8)))  float  v8f;

// ---- model dims -------------------------------------------------------------
#define BB     16
#define TT     48
#define BTN    768      // B*T
#define SCALE_ 104
#define NWT    10
#define NFEAT  8
#define HID    512
#define NH     2
#define EH     256      // head dim
#define PRED   24
#define L1C    100
#define SK1C   25
#define U1C    25
#define L2C    21
#define SK2C   12
#define U2C    12
#define LP1    128      // L1 padded to mult of 32 (ctx GEMM K)
#define LP2    32       // L2 padded
#define K1PAD  64       // conv1 im2col K (=50) padded
#define GATES  4096     // 8*HID
#define HSTATE 1024     // 2*HID
#define C12    6144     // 12*HID

// =============================================================================
// Batched WMMA GEMM:  C[M,N] = A[M,K] * B^T (+bias[N]),  B is (N,K) row-major.
// A,B bf16; fp32 accumulate; optional fp32 C and/or bf16 Cb outputs.
// Batch z: offset = (z/subB)*strideOuter + (z%subB)*strideInner  (elements).
// 256 threads = 8 waves arranged 4(M) x 2(N); each wave computes a 16x64 strip
// (1 A fragment reused across 4 B fragments -> 4 v_wmma per K step).
// K must be a multiple of 32; lda/ldb must be multiples of 8 (guaranteed by
// the padded buffer layouts below). Out-of-range rows/cols are loaded from
// clamped pointers (garbage is fine: those D entries are masked at store).
// =============================================================================
__global__ __launch_bounds__(256)
void gemm_wmma(const __bf16* __restrict__ A, long soA, long siA, int lda,
               const __bf16* __restrict__ Bm, long soB, long siB, int ldb,
               float* C, long soC, long siC, int ldc,
               __bf16* Cb, long soCb, long siCb, int ldcb,
               const float* __restrict__ bias,
               int M, int N, int Kd, int subB)
{
    const int lane = threadIdx.x & 31;
    const int wave = threadIdx.x >> 5;
    const int m0 = (blockIdx.x * 4 + (wave >> 1)) * 16;
    const int n0 = (blockIdx.y * 2 + (wave & 1)) * 64;
    if (m0 >= M || n0 >= N) return;            // wave-uniform: EXEC stays all-1s

    const long zo = blockIdx.z / subB, zi = blockIdx.z % subB;
    A  += zo * soA + zi * siA;
    Bm += zo * soB + zi * siB;

    const int half = lane >> 4;                // 0: lanes 0-15, 1: lanes 16-31
    const int sub  = lane & 15;
    const int ml = m0 + sub;

    // A fragment rows: halves [0..7] <- K half*8+0..7 ; [8..15] <- K 16+half*8+..
    const __bf16* Ap = A + (long)((ml < M) ? ml : (M - 1)) * lda + half * 8;
    // B fragment: lane = N col, [0..7] <- K half*8+0..7 ; [8..15] <- K 16+half*8+..
    const __bf16* Bp[4];
#pragma unroll
    for (int j = 0; j < 4; ++j) {
        const int nl = n0 + 16 * j + sub;
        Bp[j] = Bm + (long)((nl < N) ? nl : (N - 1)) * ldb + half * 8;
    }

    v8f acc0 = {}, acc1 = {}, acc2 = {}, acc3 = {};
    for (int k0 = 0; k0 < Kd; k0 += 32) {
        const v8bf alo = *(const v8bf*)(Ap + k0);
        const v8bf ahi = *(const v8bf*)(Ap + k0 + 16);
        const v16bf a = __builtin_shufflevector(alo, ahi,
            0, 1, 2, 3, 4, 5, 6, 7, 8, 9, 10, 11, 12, 13, 14, 15);
        if (k0 + 32 < Kd) {                    // next K tile -> global_prefetch_b8
            __builtin_prefetch(Ap + k0 + 32, 0, 3);
            __builtin_prefetch(Bp[0] + k0 + 32, 0, 3);
        }
        {
            const v8bf blo = *(const v8bf*)(Bp[0] + k0);
            const v8bf bhi = *(const v8bf*)(Bp[0] + k0 + 16);
            const v16bf b = __builtin_shufflevector(blo, bhi,
                0, 1, 2, 3, 4, 5, 6, 7, 8, 9, 10, 11, 12, 13, 14, 15);
            acc0 = __builtin_amdgcn_wmma_f32_16x16x32_bf16(false, a, false, b,
                                                           (short)0, acc0, false, false);
        }
        {
            const v8bf blo = *(const v8bf*)(Bp[1] + k0);
            const v8bf bhi = *(const v8bf*)(Bp[1] + k0 + 16);
            const v16bf b = __builtin_shufflevector(blo, bhi,
                0, 1, 2, 3, 4, 5, 6, 7, 8, 9, 10, 11, 12, 13, 14, 15);
            acc1 = __builtin_amdgcn_wmma_f32_16x16x32_bf16(false, a, false, b,
                                                           (short)0, acc1, false, false);
        }
        {
            const v8bf blo = *(const v8bf*)(Bp[2] + k0);
            const v8bf bhi = *(const v8bf*)(Bp[2] + k0 + 16);
            const v16bf b = __builtin_shufflevector(blo, bhi,
                0, 1, 2, 3, 4, 5, 6, 7, 8, 9, 10, 11, 12, 13, 14, 15);
            acc2 = __builtin_amdgcn_wmma_f32_16x16x32_bf16(false, a, false, b,
                                                           (short)0, acc2, false, false);
        }
        {
            const v8bf blo = *(const v8bf*)(Bp[3] + k0);
            const v8bf bhi = *(const v8bf*)(Bp[3] + k0 + 16);
            const v16bf b = __builtin_shufflevector(blo, bhi,
                0, 1, 2, 3, 4, 5, 6, 7, 8, 9, 10, 11, 12, 13, 14, 15);
            acc3 = __builtin_amdgcn_wmma_f32_16x16x32_bf16(false, a, false, b,
                                                           (short)0, acc3, false, false);
        }
    }

    // epilogue: C/D layout -> VGPR r holds row m0+r (lanes 0-15) / m0+8+r (16-31)
    float* Cz  = C  ? (C  + zo * soC  + zi * siC)  : nullptr;
    __bf16* Cbz = Cb ? (Cb + zo * soCb + zi * siCb) : nullptr;
#pragma unroll
    for (int j = 0; j < 4; ++j) {
        const v8f accj = (j == 0) ? acc0 : (j == 1) ? acc1 : (j == 2) ? acc2 : acc3;
        const int nl = n0 + 16 * j + sub;
        if (nl >= N) continue;
        const float bv = bias ? bias[nl] : 0.f;
#pragma unroll
        for (int r = 0; r < 8; ++r) {
            const int m = m0 + r + half * 8;
            if (m < M) {
                const float v = accj[r] + bv;
                if (Cz)  Cz [(long)m * ldc  + nl] = v;
                if (Cbz) Cbz[(long)m * ldcb + nl] = (__bf16)v;
            }
        }
    }
}

// =============================================================================
// Elementwise / helper kernels
// =============================================================================

// fp32 (rows x scols) -> bf16 (rows x dcols), zero-padding cols >= scols
__global__ void to_bf16_pad(const float* __restrict__ src, __bf16* __restrict__ dst,
                            int rows, int scols, int dcols)
{
    long tid = (long)blockIdx.x * blockDim.x + threadIdx.x;
    long total = (long)rows * dcols;
    if (tid >= total) return;
    int c = (int)(tid % dcols);
    long r = tid / dcols;
    dst[tid] = (c < scols) ? (__bf16)src[r * scols + c] : (__bf16)0.f;
}

// turbine select + positional-encoding concat + im2col for conv1
// X1: (BT*L1, K1PAD) bf16, col = c*5 + t  (c: 0-7 feat, 8 sin, 9 cos)
__global__ void im2col1_kernel(const float* __restrict__ in, const int* __restrict__ wtp,
                               __bf16* __restrict__ X)
{
    long tid = (long)blockIdx.x * blockDim.x + threadIdx.x;
    long total = (long)BTN * L1C * K1PAD;
    if (tid >= total) return;
    int col = (int)(tid & (K1PAD - 1));
    long row = tid >> 6;
    int l = (int)(row % L1C);
    long bt = row / L1C;
    __bf16 v = (__bf16)0.f;
    if (col < 50) {
        int c = col / 5, t = col % 5;
        int pos = l + t;
        float x;
        if (c < NFEAT) x = in[((bt * SCALE_ + pos) * NWT + wtp[0]) * NFEAT + c];
        else {
            float p = (float)pos / (float)SCALE_;
            x = (c == 8) ? __sinf(p) : __cosf(p);
        }
        v = (__bf16)x;
    }
    X[tid] = v;
}

// im2col for conv2: src attn1 out (BT,25,512) bf16 -> X2 (BT*21, 2560)
__global__ void im2col2_kernel(const __bf16* __restrict__ src, __bf16* __restrict__ X)
{
    long tid = (long)blockIdx.x * blockDim.x + threadIdx.x;
    long total = (long)BTN * L2C * (HID * 5);
    if (tid >= total) return;
    int col = (int)(tid % (HID * 5));
    long row = tid / (HID * 5);
    int l = (int)(row % L2C);
    long bt = row / L2C;
    int c = col / 5, t = col % 5;
    X[tid] = src[(bt * 25 + l + t) * HID + c];
}

// per-head cumsum of V over sequence, written TRANSPOSED:
// out: (BT*NH, EH, LPAD) bf16, i.e. out[bh][e][l] = cumsum_l V[bt,l,h*EH+e];
// zero-padded for l in [L, LPAD). This makes the context GEMM an A*B^T GEMM.
__global__ void cumsum_heads(const __bf16* __restrict__ V, __bf16* __restrict__ out,
                             int L, int LPAD)
{
    long tid = (long)blockIdx.x * blockDim.x + threadIdx.x;
    if (tid >= (long)BTN * NH * EH) return;
    int e = (int)(tid % EH);
    int h = (int)((tid / EH) % NH);
    long bt = tid / (EH * NH);
    const __bf16* src = V + (bt * L) * HID + h * EH + e;
    __bf16* dst = out + ((bt * NH + h) * (long)EH + e) * LPAD;
    float acc = 0.f;
    for (int l = 0; l < L; ++l) { acc += (float)src[(long)l * HID]; dst[l] = (__bf16)acc; }
    for (int l = L; l < LPAD; ++l) dst[l] = (__bf16)0.f;
}

// ProbSparse: sparsity measure from sampled scores, top-U selection, softmax
// of selected rows -> bf16 probs (U x LPAD, zero-padded cols). One block per (bt,h).
template<int L, int SK, int U, int LPAD>
__global__ void probsel_kernel(const float* __restrict__ scores, const int* __restrict__ idx,
                               __bf16* __restrict__ probs, float scale)
{
    const int bh = blockIdx.x;
    const float* S = scores + (long)bh * L * L;
    __shared__ float m[L];
    __shared__ int top[U];
    for (int l = threadIdx.x; l < L; l += blockDim.x) {
        const float* row = S + (long)l * L;
        float mx = -1e30f, sm = 0.f;
        for (int s = 0; s < SK; ++s) {
            float v = row[idx[l * SK + s]];
            mx = fmaxf(mx, v); sm += v;
        }
        m[l] = mx - sm / (float)L;
    }
    __syncthreads();
    if (threadIdx.x == 0) {
        for (int u = 0; u < U; ++u) {
            float best = -1e30f; int bi = 0;
            for (int l = 0; l < L; ++l) if (m[l] > best) { best = m[l]; bi = l; }
            top[u] = bi; m[bi] = -1e30f;
        }
    }
    __syncthreads();
    if (threadIdx.x < U) {
        const float* row = S + (long)top[threadIdx.x] * L;
        float mx = -1e30f;
        for (int l = 0; l < L; ++l) mx = fmaxf(mx, row[l] * scale);
        float sum = 0.f;
        for (int l = 0; l < L; ++l) sum += __expf(row[l] * scale - mx);
        float inv = 1.f / sum;
        __bf16* dst = probs + ((long)bh * U + threadIdx.x) * LPAD;
        for (int l = 0; l < L; ++l) dst[l] = (__bf16)(__expf(row[l] * scale - mx) * inv);
        for (int l = L; l < LPAD; ++l) dst[l] = (__bf16)0.f;
    }
}

// LayerNorm over C12 features, one block per row, bf16 output
__global__ void layernorm_kernel(const float* __restrict__ x, const float* __restrict__ g,
                                 const float* __restrict__ b, __bf16* __restrict__ out)
{
    const int row = blockIdx.x;
    const float* xr = x + (long)row * C12;
    __shared__ float s1[256], s2[256];
    float a = 0.f, q = 0.f;
    for (int i = threadIdx.x; i < C12; i += 256) { float v = xr[i]; a += v; q += v * v; }
    s1[threadIdx.x] = a; s2[threadIdx.x] = q;
    __syncthreads();
    for (int o = 128; o > 0; o >>= 1) {
        if (threadIdx.x < o) { s1[threadIdx.x] += s1[threadIdx.x + o]; s2[threadIdx.x] += s2[threadIdx.x + o]; }
        __syncthreads();
    }
    float mean = s1[0] / (float)C12;
    float var  = s2[0] / (float)C12 - mean * mean;
    float inv  = rsqrtf(var + 1e-5f);
    for (int i = threadIdx.x; i < C12; i += 256)
        out[(long)row * C12 + i] = (__bf16)((xr[i] - mean) * inv * g[i] + b[i]);
}

__global__ void add_vec(const float* a, const float* b, float* o, int n)
{
    int i = blockIdx.x * blockDim.x + threadIdx.x;
    if (i < n) o[i] = a[i] + b[i];
}

__global__ void lstm_init(float* h, float* c, __bf16* hb)
{
    int i = blockIdx.x * blockDim.x + threadIdx.x;
    if (i < BB * HSTATE) { h[i] = 0.f; c[i] = 0.f; hb[i] = (__bf16)0.f; }
}

// one LSTM timestep pointwise (gate order i,f,g,o)
__global__ void lstm_step(const float* __restrict__ gx, const float* __restrict__ gh,
                          float* __restrict__ c, float* __restrict__ h,
                          __bf16* __restrict__ hb, int t)
{
    int tid = blockIdx.x * blockDim.x + threadIdx.x;
    if (tid >= BB * HSTATE) return;
    int b = tid >> 10, j = tid & (HSTATE - 1);
    long rx = (long)(b * TT + t) * GATES;
    long rh = (long)b * GATES;
    float gi = gx[rx + j]                + gh[rh + j];
    float gf = gx[rx + HSTATE + j]       + gh[rh + HSTATE + j];
    float gg = gx[rx + 2 * HSTATE + j]   + gh[rh + 2 * HSTATE + j];
    float go = gx[rx + 3 * HSTATE + j]   + gh[rh + 3 * HSTATE + j];
    float si = 1.f / (1.f + __expf(-gi));
    float sf = 1.f / (1.f + __expf(-gf));
    float so = 1.f / (1.f + __expf(-go));
    float cc = sf * c[tid] + si * tanhf(gg);
    float hh = so * tanhf(cc);
    c[tid] = cc; h[tid] = hh; hb[tid] = (__bf16)hh;
}

// =============================================================================
// host
// =============================================================================
static inline dim3 gemm_grid(int M, int N, int batch)
{
    return dim3((unsigned)((M + 63) / 64), (unsigned)((N + 127) / 128), (unsigned)batch);
}
static inline dim3 ew_grid(long total) { return dim3((unsigned)((total + 255) / 256)); }

extern "C" void kernel_launch(void* const* d_in, const int* in_sizes, int n_in,
                              void* d_out, int out_size, void* d_ws, size_t ws_size,
                              hipStream_t stream)
{
    (void)in_sizes; (void)n_in; (void)out_size; (void)ws_size;
    const float* input   = (const float*)d_in[0];
    const int*   wtp     = (const int*)  d_in[2];
    const int*   idx1    = (const int*)  d_in[3];
    const int*   idx2    = (const int*)  d_in[4];
    const float* conv1_w = (const float*)d_in[5];
    const float* conv1_b = (const float*)d_in[6];
    const float* conv2_w = (const float*)d_in[7];
    const float* conv2_b = (const float*)d_in[8];
    const float* a1w[4] = { (const float*)d_in[9],  (const float*)d_in[11],
                            (const float*)d_in[13], (const float*)d_in[15] };
    const float* a1b[4] = { (const float*)d_in[10], (const float*)d_in[12],
                            (const float*)d_in[14], (const float*)d_in[16] };
    const float* a2w[4] = { (const float*)d_in[17], (const float*)d_in[19],
                            (const float*)d_in[21], (const float*)d_in[23] };
    const float* a2b[4] = { (const float*)d_in[18], (const float*)d_in[20],
                            (const float*)d_in[22], (const float*)d_in[24] };
    const float* ln_g     = (const float*)d_in[25];
    const float* ln_b     = (const float*)d_in[26];
    const float* lstm_wih = (const float*)d_in[27];
    const float* lstm_whh = (const float*)d_in[28];
    const float* lstm_bih = (const float*)d_in[29];
    const float* lstm_bhh = (const float*)d_in[30];
    const float* fin_w    = (const float*)d_in[31];
    const float* fin_b    = (const float*)d_in[32];
    float* out = (float*)d_out;

    // ---- workspace bump allocator (256B aligned -> b128-friendly) ----
    char* base = (char*)d_ws;
    size_t cur = 0;
    auto alloc = [&](size_t bytes) -> void* {
        void* p = base + cur;
        cur += (bytes + 255) & ~(size_t)255;
        return p;
    };
    auto ab = [&](long elems) { return (__bf16*)alloc((size_t)elems * 2); };
    auto af = [&](long elems) { return (float*)alloc((size_t)elems * 4); };

    __bf16* W1b   = ab(512L * K1PAD);
    __bf16* X1    = ab((long)BTN * L1C * K1PAD);
    __bf16* H1b   = ab((long)BTN * L1C * HID);
    __bf16* wA1[4]; for (int i = 0; i < 4; ++i) wA1[i] = ab(512L * 512);
    __bf16* wA2[4]; for (int i = 0; i < 4; ++i) wA2[i] = ab(512L * 512);
    __bf16* Q1b   = ab((long)BTN * L1C * HID);
    __bf16* K1b   = ab((long)BTN * L1C * HID);
    __bf16* V1b   = ab((long)BTN * L1C * HID);
    __bf16* vcum1 = ab((long)BTN * NH * EH * LP1);   // transposed (bh, e, l)
    float*  sc1   = af((long)BTN * NH * L1C * L1C);
    __bf16* pr1   = ab((long)BTN * NH * U1C * LP1);
    __bf16* ctx1  = ab((long)BTN * NH * U1C * EH);
    __bf16* att1  = ab((long)BTN * 25 * HID);
    __bf16* X2    = ab((long)BTN * L2C * HID * 5);
    __bf16* W2b   = ab(512L * HID * 5);
    __bf16* H2b   = ab((long)BTN * L2C * HID);
    __bf16* Q2b   = ab((long)BTN * L2C * HID);
    __bf16* K2b   = ab((long)BTN * L2C * HID);
    __bf16* V2b   = ab((long)BTN * L2C * HID);
    __bf16* vcum2 = ab((long)BTN * NH * EH * LP2);   // transposed (bh, e, l)
    float*  sc2   = af((long)BTN * NH * L2C * L2C);
    __bf16* pr2   = ab((long)BTN * NH * U2C * LP2);
    __bf16* ctx2  = ab((long)BTN * NH * U2C * EH);
    float*  hpre  = af((long)BTN * C12);
    __bf16* hlnb  = ab((long)BTN * C12);
    __bf16* wihb  = ab((long)GATES * C12);
    __bf16* whhb  = ab((long)GATES * HSTATE);
    __bf16* finb  = ab((long)PRED * HSTATE);
    float*  biasg = af(GATES);
    float*  gx    = af((long)BTN * GATES);
    float*  gh    = af((long)BB * GATES);
    float*  hst   = af(BB * HSTATE);
    float*  cst   = af(BB * HSTATE);
    __bf16* hb    = ab(BB * HSTATE);

    // ---- weight conversions (fp32 -> bf16, padded where needed) ----
    to_bf16_pad<<<ew_grid(512L * K1PAD), 256, 0, stream>>>(conv1_w, W1b, 512, 50, K1PAD);
    to_bf16_pad<<<ew_grid(512L * 2560), 256, 0, stream>>>(conv2_w, W2b, 512, 2560, 2560);
    for (int i = 0; i < 4; ++i) {
        to_bf16_pad<<<ew_grid(512L * 512), 256, 0, stream>>>(a1w[i], wA1[i], 512, 512, 512);
        to_bf16_pad<<<ew_grid(512L * 512), 256, 0, stream>>>(a2w[i], wA2[i], 512, 512, 512);
    }
    to_bf16_pad<<<ew_grid((long)GATES * C12), 256, 0, stream>>>(lstm_wih, wihb, GATES, C12, C12);
    to_bf16_pad<<<ew_grid((long)GATES * HSTATE), 256, 0, stream>>>(lstm_whh, whhb, GATES, HSTATE, HSTATE);
    to_bf16_pad<<<ew_grid((long)PRED * HSTATE), 256, 0, stream>>>(fin_w, finb, PRED, HSTATE, HSTATE);
    add_vec<<<ew_grid(GATES), 256, 0, stream>>>(lstm_bih, lstm_bhh, biasg, GATES);

    // ---- stage 1: conv1 (im2col GEMM) ----
    im2col1_kernel<<<ew_grid((long)BTN * L1C * K1PAD), 256, 0, stream>>>(input, wtp, X1);
    gemm_wmma<<<gemm_grid(BTN * L1C, HID, 1), 256, 0, stream>>>(
        X1, 0, 0, K1PAD, W1b, 0, 0, K1PAD,
        nullptr, 0, 0, 0, H1b, 0, 0, HID, conv1_b,
        BTN * L1C, HID, K1PAD, 1);

    // ---- attention 1 ----
    const int M1 = BTN * L1C;
    gemm_wmma<<<gemm_grid(M1, HID, 1), 256, 0, stream>>>(H1b, 0, 0, HID, wA1[0], 0, 0, HID,
        nullptr, 0, 0, 0, Q1b, 0, 0, HID, a1b[0], M1, HID, HID, 1);
    gemm_wmma<<<gemm_grid(M1, HID, 1), 256, 0, stream>>>(H1b, 0, 0, HID, wA1[1], 0, 0, HID,
        nullptr, 0, 0, 0, K1b, 0, 0, HID, a1b[1], M1, HID, HID, 1);
    gemm_wmma<<<gemm_grid(M1, HID, 1), 256, 0, stream>>>(H1b, 0, 0, HID, wA1[2], 0, 0, HID,
        nullptr, 0, 0, 0, V1b, 0, 0, HID, a1b[2], M1, HID, HID, 1);
    cumsum_heads<<<ew_grid((long)BTN * NH * EH), 256, 0, stream>>>(V1b, vcum1, L1C, LP1);
    // scores[bh] = Q[bh] (100x256) * K[bh]^T ; bh = bt*2+h
    gemm_wmma<<<gemm_grid(L1C, L1C, BTN * NH), 256, 0, stream>>>(
        Q1b, (long)L1C * HID, EH, HID, K1b, (long)L1C * HID, EH, HID,
        sc1, 2L * L1C * L1C, (long)L1C * L1C, L1C, nullptr, 0, 0, 0, nullptr,
        L1C, L1C, EH, NH);
    probsel_kernel<L1C, SK1C, U1C, LP1><<<BTN * NH, 128, 0, stream>>>(sc1, idx1, pr1, 0.0625f);
    // ctx[bh] = probs (25 x LP1) * vcumT (256 x LP1)^T
    gemm_wmma<<<gemm_grid(U1C, EH, BTN * NH), 256, 0, stream>>>(
        pr1, 2L * U1C * LP1, (long)U1C * LP1, LP1, vcum1, 2L * EH * LP1, (long)EH * LP1, LP1,
        nullptr, 0, 0, 0, ctx1, 2L * U1C * EH, (long)U1C * EH, EH, nullptr,
        U1C, EH, LP1, NH);
    // output projection: ctx viewed as (BT*25, 512)
    gemm_wmma<<<gemm_grid(BTN * 25, HID, 1), 256, 0, stream>>>(
        ctx1, 0, 0, HID, wA1[3], 0, 0, HID,
        nullptr, 0, 0, 0, att1, 0, 0, HID, a1b[3], BTN * 25, HID, HID, 1);

    // ---- stage 2: conv2 ----
    im2col2_kernel<<<ew_grid((long)BTN * L2C * HID * 5), 256, 0, stream>>>(att1, X2);
    gemm_wmma<<<gemm_grid(BTN * L2C, HID, 1), 256, 0, stream>>>(
        X2, 0, 0, HID * 5, W2b, 0, 0, HID * 5,
        nullptr, 0, 0, 0, H2b, 0, 0, HID, conv2_b, BTN * L2C, HID, HID * 5, 1);

    // ---- attention 2 ----
    const int M2 = BTN * L2C;
    gemm_wmma<<<gemm_grid(M2, HID, 1), 256, 0, stream>>>(H2b, 0, 0, HID, wA2[0], 0, 0, HID,
        nullptr, 0, 0, 0, Q2b, 0, 0, HID, a2b[0], M2, HID, HID, 1);
    gemm_wmma<<<gemm_grid(M2, HID, 1), 256, 0, stream>>>(H2b, 0, 0, HID, wA2[1], 0, 0, HID,
        nullptr, 0, 0, 0, K2b, 0, 0, HID, a2b[1], M2, HID, HID, 1);
    gemm_wmma<<<gemm_grid(M2, HID, 1), 256, 0, stream>>>(H2b, 0, 0, HID, wA2[2], 0, 0, HID,
        nullptr, 0, 0, 0, V2b, 0, 0, HID, a2b[2], M2, HID, HID, 1);
    cumsum_heads<<<ew_grid((long)BTN * NH * EH), 256, 0, stream>>>(V2b, vcum2, L2C, LP2);
    gemm_wmma<<<gemm_grid(L2C, L2C, BTN * NH), 256, 0, stream>>>(
        Q2b, (long)L2C * HID, EH, HID, K2b, (long)L2C * HID, EH, HID,
        sc2, 2L * L2C * L2C, (long)L2C * L2C, L2C, nullptr, 0, 0, 0, nullptr,
        L2C, L2C, EH, NH);
    probsel_kernel<L2C, SK2C, U2C, LP2><<<BTN * NH, 128, 0, stream>>>(sc2, idx2, pr2, 0.0625f);
    gemm_wmma<<<gemm_grid(U2C, EH, BTN * NH), 256, 0, stream>>>(
        pr2, 2L * U2C * LP2, (long)U2C * LP2, LP2, vcum2, 2L * EH * LP2, (long)EH * LP2, LP2,
        nullptr, 0, 0, 0, ctx2, 2L * U2C * EH, (long)U2C * EH, EH, nullptr,
        U2C, EH, LP2, NH);
    // output projection, fp32 out for LayerNorm: (BT*12, 512) == (BT, 6144)
    gemm_wmma<<<gemm_grid(BTN * 12, HID, 1), 256, 0, stream>>>(
        ctx2, 0, 0, HID, wA2[3], 0, 0, HID,
        hpre, 0, 0, HID, nullptr, 0, 0, 0, a2b[3], BTN * 12, HID, HID, 1);

    // ---- LayerNorm ----
    layernorm_kernel<<<BTN, 256, 0, stream>>>(hpre, ln_g, ln_b, hlnb);

    // ---- LSTM: batched input projection then 48 sequential steps ----
    gemm_wmma<<<gemm_grid(BTN, GATES, 1), 256, 0, stream>>>(
        hlnb, 0, 0, C12, wihb, 0, 0, C12,
        gx, 0, 0, GATES, nullptr, 0, 0, 0, biasg, BTN, GATES, C12, 1);
    lstm_init<<<ew_grid(BB * HSTATE), 256, 0, stream>>>(hst, cst, hb);
    for (int t = 0; t < TT; ++t) {
        gemm_wmma<<<gemm_grid(BB, GATES, 1), 256, 0, stream>>>(
            hb, 0, 0, HSTATE, whhb, 0, 0, HSTATE,
            gh, 0, 0, GATES, nullptr, 0, 0, 0, nullptr, BB, GATES, HSTATE, 1);
        lstm_step<<<ew_grid(BB * HSTATE), 256, 0, stream>>>(gx, gh, cst, hst, hb, t);
    }

    // ---- final linear -> d_out (16, 24) fp32 ----
    gemm_wmma<<<gemm_grid(BB, PRED, 1), 256, 0, stream>>>(
        hb, 0, 0, HSTATE, finb, 0, 0, HSTATE,
        out, 0, 0, PRED, nullptr, 0, 0, 0, fin_b, BB, PRED, HSTATE, 1);
}
